// TinyVectorQuantizer_77695958385403
// MI455X (gfx1250) — compile-verified, matrix operand order checked by reading, not run on hardware
//
#include <hip/hip_runtime.h>
#include <hip/hip_bf16.h>

typedef __attribute__((ext_vector_type(16))) __bf16 v16bf;
typedef __attribute__((ext_vector_type(8)))  __bf16 v8bf;
typedef __attribute__((ext_vector_type(8)))  float  v8f;
typedef __attribute__((ext_vector_type(4)))  float  v4f;

// 16-byte unit type matching the async-to-LDS builtin's prototype
typedef int i128v __attribute__((vector_size(16)));
typedef __attribute__((address_space(1))) i128v i128_g;   // global AS1
typedef __attribute__((address_space(3))) i128v i128_l;   // LDS AS3

#define D_DIM   128
#define T_DIM   2048
#define K_CODES 1024
#define B_DIM   16
#define N_TOK   (B_DIM * T_DIM)       // 32768 tokens
#define N_TILES (N_TOK / 16)          // 2048 token tiles
#define WPB     8                     // waves per block
#define BLOCK_T (WPB * 32)            // 256 threads
#define N_BLOCKS (N_TILES / WPB)      // 256 blocks

// ---- dynamic LDS layout (bytes) -------------------------------------------
// xs   : [WPB][128][16] f32   x tiles, one per wave                 65536
// buf  : [2][256][128] bf16   double-buffered codebook chunk       131072
// idx_s: [WPB][16] int                                                512
// lred : [WPB] float                                                   32
// enb  : [2][256] f32         double-buffered chunk norms            2048
#define LDS_XS_BYTES  (WPB * 128 * 16 * 4)
#define LDS_BUF_OFF   LDS_XS_BYTES
#define LDS_BUF_HALF  (256 * 128 * 2)               // 64 KB per chunk buffer
#define LDS_IDX_OFF   (LDS_BUF_OFF + 2 * LDS_BUF_HALF)
#define LDS_LRED_OFF  (LDS_IDX_OFF + WPB * 16 * 4)
#define LDS_ENB_OFF   (LDS_LRED_OFF + WPB * 4)
#define LDS_TOTAL     (LDS_ENB_OFF + 2 * 256 * 4)   // 199200 B (< 320 KB/WGP)

// CDNA5 async global->LDS copy (ASYNCcnt path). Builtin confirmed present.
__device__ __forceinline__ void async_copy16(void* lds_dst, const void* gsrc) {
#if __has_builtin(__builtin_amdgcn_global_load_async_to_lds_b128)
    __builtin_amdgcn_global_load_async_to_lds_b128(
        (i128_g*)(uintptr_t)gsrc,
        (i128_l*)(unsigned int)(uintptr_t)lds_dst, 0, 0);
#else
    unsigned int l = (unsigned int)(uintptr_t)lds_dst;
    asm volatile("global_load_async_to_lds_b128 %0, %1, off"
                 :: "v"(l), "v"(gsrc) : "memory");
#endif
}

__device__ __forceinline__ void wait_async0() {
#if __has_builtin(__builtin_amdgcn_s_wait_asynccnt)
    __builtin_amdgcn_s_wait_asynccnt(0);
#else
    asm volatile("s_wait_asynccnt 0x0" ::: "memory");
#endif
}

// ---------------------------------------------------------------------------
// Kernel 1: per-code squared norms + bf16 copy of the codebook.
// ---------------------------------------------------------------------------
__global__ void vq_prep(const float* __restrict__ emb,
                        float* __restrict__ enorm,
                        __bf16* __restrict__ ebf) {
    const int k = blockIdx.x;
    const int d = threadIdx.x;
    float e = emb[(size_t)k * D_DIM + d];
    ebf[(size_t)k * D_DIM + d] = (__bf16)e;
    float s = e * e;
    #pragma unroll
    for (int m = 1; m < 32; m <<= 1) s += __shfl_xor(s, m, 32);
    __shared__ float r[4];
    if ((d & 31) == 0) r[d >> 5] = s;
    __syncthreads();
    if (d == 0) enorm[k] = r[0] + r[1] + r[2] + r[3];
}

// ---------------------------------------------------------------------------
// Kernel 2: fused distance-GEMM (bf16 WMMA) + argmin + gather + loss partials.
// Block = 8 waves, one 16-token tile per wave. Codebook (+norms) staged into
// LDS in 4 double-buffered 64 KB chunks via async-to-LDS, shared by 8 waves.
// Hot loop: 4 code tiles per step -> 32 hoisted ds_load_b128 feeding 16 WMMAs
// in 4 independent accumulator chains (covers the WMMA->WMMA RAW hazard).
//
// WMMA 16x16x32 bf16 operand layout (ISA 7.12.2 16-bit A table, B mirrored):
//   lane L: row/col = L&15, khalf = L>>4
//   fragment element s (0..15): k = (s>=8 ? 16 : 0) + khalf*8 + (s&7)
// C/D: vgpr j, lane L -> M = j + 8*(L>>4), N = L&15.
// Codebook rows in LDS use a 16B XOR swizzle: pcol = col16 ^ (row & 15), which
// makes the per-lane ds_read_b128 of 16 distinct rows bank-conflict-free.
// ---------------------------------------------------------------------------
__global__ __launch_bounds__(BLOCK_T)
void vq_main(const float* __restrict__ x,
             const float* __restrict__ emb,
             const float* __restrict__ enorm,
             const __bf16* __restrict__ ebf,
             float* __restrict__ out_q,
             float* __restrict__ out_idx,
             float* __restrict__ partial) {
    extern __shared__ char smem[];
    float* xs    = (float*)smem;                       // [WPB][128][16]
    char*  buf   = smem + LDS_BUF_OFF;                 // [2][256 rows][256 B]
    int*   idx_s = (int*)(smem + LDS_IDX_OFF);         // [WPB][16]
    float* lred  = (float*)(smem + LDS_LRED_OFF);      // [WPB]
    char*  enb   = smem + LDS_ENB_OFF;                 // [2][256] f32

    const int tid  = threadIdx.x;
    const int w    = tid >> 5;
    const int L    = tid & 31;
    const int tile = blockIdx.x * WPB + w;
    const int bb   = tile / (T_DIM / 16);
    const int t0   = (tile % (T_DIM / 16)) * 16;

    // ---- stage x tile (16 t x 128 d f32) into LDS via async copies ----
    const float* xg = x + (size_t)bb * D_DIM * T_DIM + t0;
    #pragma unroll 4
    for (int it = 0; it < 16; ++it) {
        const int d  = it * 8 + (L >> 2);
        const int tq = (L & 3) * 4;
        async_copy16(&xs[(w * 128 + d) * 16 + tq], xg + (size_t)d * T_DIM + tq);
    }

    // ---- stage codebook chunk 0 (codes 0..255) + its norms into buf[0] ----
    {
        const char* src = (const char*)ebf;            // chunk 0 base
        #pragma unroll 4
        for (int u = 0; u < 16; ++u) {
            const int unit = u * BLOCK_T + tid;        // 0..4095
            const int row  = unit >> 4;
            const int pcol = (unit & 15) ^ (row & 15);
            async_copy16(buf + row * 256 + (pcol << 4), src + unit * 16);
        }
        if (tid < 64)
            async_copy16(enb + tid * 16, (const char*)enorm + tid * 16);
    }
    wait_async0();
    __syncthreads();

    const int hi   = L >> 4;     // K-half select
    const int mrow = L & 15;     // A: token row / B: code column / C: N

    // ---- build A fragments once (reused against all 1024 codes) ----
    v16bf afrag[4];
    #pragma unroll
    for (int ks = 0; ks < 4; ++ks) {
        #pragma unroll
        for (int s = 0; s < 16; ++s) {
            const int k = ((s >= 8) ? 16 : 0) + hi * 8 + (s & 7);
            afrag[ks][s] = (__bf16)xs[(w * 128 + ks * 32 + k) * 16 + mrow];
        }
    }

    float minv[8];
    int   mini[8];
    #pragma unroll
    for (int j = 0; j < 8; ++j) { minv[j] = 3.4e38f; mini[j] = 0; }

    // ---- 4 chunks x 16 code tiles; double-buffered; 4 acc chains for ILP ----
    for (int c = 0; c < 4; ++c) {
        const char*  bufc = buf + (c & 1) * LDS_BUF_HALF;
        const float* enc  = (const float*)(enb + (c & 1) * 1024);
        if (c + 1 < 4) {   // prefetch next chunk into the other buffer
            char* dst = buf + ((c + 1) & 1) * LDS_BUF_HALF;
            const char* src = (const char*)(ebf + (size_t)(c + 1) * 256 * D_DIM);
            #pragma unroll 4
            for (int u = 0; u < 16; ++u) {
                const int unit = u * BLOCK_T + tid;
                const int row  = unit >> 4;
                const int pcol = (unit & 15) ^ (row & 15);
                async_copy16(dst + row * 256 + (pcol << 4), src + unit * 16);
            }
            if (tid < 64)
                async_copy16(enb + ((c + 1) & 1) * 1024 + tid * 16,
                             (const char*)(enorm + (c + 1) * 256) + tid * 16);
        }

        const int cbase = c * 256;
        for (int ctl = 0; ctl < 16; ctl += 4) {
            // preload all B fragments for 4 code tiles (32 ds_load_b128)
            v16bf f[16];
            #pragma unroll
            for (int p = 0; p < 4; ++p) {
                const char* r = bufc + ((ctl + p) * 16 + mrow) * 256;
                #pragma unroll
                for (int ks = 0; ks < 4; ++ks) {
                    v8bf lo = *(const v8bf*)(r + (((ks * 4 + hi)     ^ mrow) << 4));
                    v8bf hh = *(const v8bf*)(r + (((ks * 4 + hi + 2) ^ mrow) << 4));
                    #pragma unroll
                    for (int s = 0; s < 8; ++s) {
                        f[p * 4 + ks][s]     = lo[s];
                        f[p * 4 + ks][8 + s] = hh[s];
                    }
                }
            }
            // 16 WMMAs, round-robin over 4 independent accumulator chains
            v8f acc[4] = {};
            #pragma unroll
            for (int ks = 0; ks < 4; ++ks) {
                #pragma unroll
                for (int p = 0; p < 4; ++p) {
                    acc[p] = __builtin_amdgcn_wmma_f32_16x16x32_bf16(
                        false, afrag[ks], false, f[p * 4 + ks],
                        (short)0, acc[p], false, false);
                }
            }
            // fused argmin update (||x||^2 is argmin-invariant)
            #pragma unroll
            for (int p = 0; p < 4; ++p) {
                const int   code = cbase + (ctl + p) * 16 + mrow;
                const float en   = enc[(ctl + p) * 16 + mrow];
                #pragma unroll
                for (int j = 0; j < 8; ++j) {
                    const float dv = en - 2.0f * acc[p][j];
                    if (dv < minv[j]) { minv[j] = dv; mini[j] = code; }
                }
            }
        }
        wait_async0();       // our async stores to the other buffer landed
        __syncthreads();     // whole block done reading bufc / writing next
    }

    // ---- reduce argmin across the 16 N-lanes of each half-wave ----
    #pragma unroll
    for (int m = 1; m < 16; m <<= 1) {
        #pragma unroll
        for (int j = 0; j < 8; ++j) {
            const float ov = __shfl_xor(minv[j], m, 32);
            const int   oi = __shfl_xor(mini[j], m, 32);
            if (ov < minv[j] || (ov == minv[j] && oi < mini[j])) {
                minv[j] = ov; mini[j] = oi;
            }
        }
    }
    if (mrow == 0) {  // lanes 0 and 16 hold rows hi*8 + j
        #pragma unroll
        for (int j = 0; j < 8; ++j) idx_s[w * 16 + hi * 8 + j] = mini[j];
    }
    __syncthreads();

    // ---- indices output (as float; value-exact) ----
    if (L < 16) out_idx[(size_t)bb * T_DIM + t0 + L] = (float)idx_s[w * 16 + L];

    // ---- gather winning rows (f32 codebook), write output, loss partial ----
    const int    tloc  = mrow;
    const int    myidx = idx_s[w * 16 + tloc];
    const float* erow  = emb + (size_t)myidx * D_DIM;
    float* oq = out_q + (size_t)bb * D_DIM * T_DIM + t0;
    float lacc = 0.0f;
    #pragma unroll 8
    for (int dd = 0; dd < 64; ++dd) {
        const int d = dd * 2 + hi;                 // half-wave: 16 contiguous t
        const float q = erow[d];
        oq[(size_t)d * T_DIM + tloc] = q;
        const float df = xs[(w * 128 + d) * 16 + tloc] - q;
        lacc += df * df;
    }
    #pragma unroll
    for (int m = 1; m < 32; m <<= 1) lacc += __shfl_xor(lacc, m, 32);
    if (L == 0) lred[w] = lacc;
    __syncthreads();
    if (tid == 0) {
        float s = 0.0f;
        #pragma unroll
        for (int i = 0; i < WPB; ++i) s += lred[i];
        partial[blockIdx.x] = s;
    }
}

// ---------------------------------------------------------------------------
// Kernel 3: deterministic fixed-order reduction of 256 block partials.
// ---------------------------------------------------------------------------
__global__ void vq_fin(const float* __restrict__ partial,
                       float* __restrict__ out_loss) {
    __shared__ float s[256];
    const int t = threadIdx.x;
    s[t] = partial[t];
    __syncthreads();
    for (int off = 128; off > 0; off >>= 1) {
        if (t < off) s[t] += s[t + off];
        __syncthreads();
    }
    if (t == 0)
        out_loss[0] = 0.25f * s[0] / (float)((size_t)B_DIM * D_DIM * T_DIM);
}

// ---------------------------------------------------------------------------
extern "C" void kernel_launch(void* const* d_in, const int* in_sizes, int n_in,
                              void* d_out, int out_size, void* d_ws, size_t ws_size,
                              hipStream_t stream) {
    (void)in_sizes; (void)n_in; (void)out_size; (void)ws_size;
    const float* x   = (const float*)d_in[0];   // (16, 128, 2048)
    const float* emb = (const float*)d_in[1];   // (1024, 128)

    float* out_q    = (float*)d_out;                                  // B*D*T
    float* out_idx  = (float*)d_out + (size_t)B_DIM * D_DIM * T_DIM;  // B*T
    float* out_loss = out_idx + (size_t)B_DIM * T_DIM;                // 1

    char* ws = (char*)d_ws;
    float*  enorm   = (float*)ws;                                     //  4 KB
    __bf16* ebf     = (__bf16*)(ws + 4096);                           // 256 KB
    float*  partial = (float*)(ws + 4096 + (size_t)K_CODES * D_DIM * 2);

    vq_prep<<<K_CODES, D_DIM, 0, stream>>>(emb, enorm, ebf);
    vq_main<<<N_BLOCKS, BLOCK_T, LDS_TOTAL, stream>>>(x, emb, enorm, ebf,
                                                      out_q, out_idx, partial);
    vq_fin<<<1, 256, 0, stream>>>(partial, out_loss);
}
